// ForwardModelTransformer_1056561954887
// MI455X (gfx1250) — matrix operand order repaired and checked by reading, not compiled, not run on hardware
//
#include <hip/hip_runtime.h>

// ---------------------------------------------------------------------------
// Model dims (compile-time constants)
// ---------------------------------------------------------------------------
#define Bc   4
#define Tc   512
#define Hc   768
#define NHc  12
#define Lc   6
#define Sc   1536           // 3*T
#define DHc  64
#define Mrows 6144          // B*S
#define FFc  3072
#define ODc  64
#define KCB  1024

typedef __attribute__((ext_vector_type(16))) __bf16  v16bf;
typedef __attribute__((ext_vector_type(8)))  float   v8f;

union BFrag { v16bf v; unsigned short u[16]; uint4 q[2]; };

static __device__ __forceinline__ unsigned short f2bf(float f) {
    unsigned int u = __float_as_uint(f);
    return (unsigned short)((u + 0x7FFFu + ((u >> 16) & 1u)) >> 16);
}
static __device__ __forceinline__ unsigned int pack2bf(float lo, float hi) {
    return (unsigned int)f2bf(lo) | ((unsigned int)f2bf(hi) << 16);
}
static __device__ __forceinline__ float gelu_exact(float x) {
    return 0.5f * x * (1.0f + erff(x * 0.70710678118654752f));
}
static __device__ __forceinline__ v8f wmma_bf16(const BFrag& a, const BFrag& b, v8f c) {
    return __builtin_amdgcn_wmma_f32_16x16x32_bf16(false, a.v, false, b.v,
                                                   (short)0, c, false, false);
}

// ---------------------------------------------------------------------------
// Embedding + VQ kernel: one block per (b,t) token
// ---------------------------------------------------------------------------
__global__ __launch_bounds__(256) void embed_vq_kernel(
    const int* __restrict__ timesteps,
    const float* __restrict__ states, const float* __restrict__ actions,
    const float* __restrict__ dynamics,
    const float* __restrict__ te,
    const float* __restrict__ W_es, const float* __restrict__ b_es,
    const float* __restrict__ W_ea, const float* __restrict__ b_ea,
    const float* __restrict__ W_ed, const float* __restrict__ b_ed,
    const float* __restrict__ cb_s, const float* __restrict__ cb_a,
    const float* __restrict__ cb_d,
    float* __restrict__ hraw, float* __restrict__ partial)
{
    const int token = blockIdx.x;
    const int b = token >> 9;       // /T (T=512)
    const int t = token & 511;
    const int tid = threadIdx.x;

    __shared__ float xs[64], xa[2], xd[5];
    __shared__ float qs[64], qa[2], qd[5];
    __shared__ float rd[256];
    __shared__ int   ri[256];
    __shared__ int   idxs[3];

    if (tid < 64) xs[tid] = states[token * 64 + tid];
    if (tid < 2)  xa[tid] = actions[token * 2 + tid];
    if (tid < 5)  xd[tid] = dynamics[token * 5 + tid];
    __syncthreads();

    // ---- VQ over states (dim 64) ----
    {
        float best = 3.4e38f; int bi = 0;
        for (int j = 0; j < 4; ++j) {
            const int row = tid + j * 256;
            const float* c = cb_s + row * 64;
            float dot = 0.f, nn = 0.f;
            for (int i = 0; i < 64; ++i) { const float cv = c[i]; dot += cv * xs[i]; nn += cv * cv; }
            const float d = nn - 2.f * dot;
            if (d < best) { best = d; bi = row; }
        }
        rd[tid] = best; ri[tid] = bi; __syncthreads();
        for (int off = 128; off > 0; off >>= 1) {
            if (tid < off) {
                if (rd[tid + off] < rd[tid] ||
                    (rd[tid + off] == rd[tid] && ri[tid + off] < ri[tid])) {
                    rd[tid] = rd[tid + off]; ri[tid] = ri[tid + off];
                }
            }
            __syncthreads();
        }
        if (tid == 0) idxs[0] = ri[0];
        __syncthreads();
    }
    // ---- VQ over actions (dim 2) ----
    {
        float best = 3.4e38f; int bi = 0;
        for (int j = 0; j < 4; ++j) {
            const int row = tid + j * 256;
            const float c0 = cb_a[row * 2], c1 = cb_a[row * 2 + 1];
            const float d = c0 * c0 + c1 * c1 - 2.f * (c0 * xa[0] + c1 * xa[1]);
            if (d < best) { best = d; bi = row; }
        }
        rd[tid] = best; ri[tid] = bi; __syncthreads();
        for (int off = 128; off > 0; off >>= 1) {
            if (tid < off) {
                if (rd[tid + off] < rd[tid] ||
                    (rd[tid + off] == rd[tid] && ri[tid + off] < ri[tid])) {
                    rd[tid] = rd[tid + off]; ri[tid] = ri[tid + off];
                }
            }
            __syncthreads();
        }
        if (tid == 0) idxs[1] = ri[0];
        __syncthreads();
    }
    // ---- VQ over dynamics (dim 5) ----
    {
        float best = 3.4e38f; int bi = 0;
        for (int j = 0; j < 4; ++j) {
            const int row = tid + j * 256;
            const float* c = cb_d + row * 5;
            float dot = 0.f, nn = 0.f;
            for (int i = 0; i < 5; ++i) { const float cv = c[i]; dot += cv * xd[i]; nn += cv * cv; }
            const float d = nn - 2.f * dot;
            if (d < best) { best = d; bi = row; }
        }
        rd[tid] = best; ri[tid] = bi; __syncthreads();
        for (int off = 128; off > 0; off >>= 1) {
            if (tid < off) {
                if (rd[tid + off] < rd[tid] ||
                    (rd[tid + off] == rd[tid] && ri[tid + off] < ri[tid])) {
                    rd[tid] = rd[tid + off]; ri[tid] = ri[tid + off];
                }
            }
            __syncthreads();
        }
        if (tid == 0) idxs[2] = ri[0];
        __syncthreads();
    }

    if (tid < 64) qs[tid] = cb_s[idxs[0] * 64 + tid];
    if (tid < 2)  qa[tid] = cb_a[idxs[1] * 2 + tid];
    if (tid < 5)  qd[tid] = cb_d[idxs[2] * 5 + tid];
    __syncthreads();

    if (tid == 0) {
        float cs = 0.f, ca = 0.f, cd = 0.f;
        for (int i = 0; i < 64; ++i) { const float d = qs[i] - xs[i]; cs += d * d; }
        for (int i = 0; i < 2;  ++i) { const float d = qa[i] - xa[i]; ca += d * d; }
        for (int i = 0; i < 5;  ++i) { const float d = qd[i] - xd[i]; cd += d * d; }
        partial[token * 3 + 0] = cs;
        partial[token * 3 + 1] = ca;
        partial[token * 3 + 2] = cd;
    }

    const int ts = timesteps[token];
    const float* terow = te + (long)ts * Hc;
    const long base = ((long)b * Sc + 3 * t) * Hc;
    for (int j = tid; j < Hc; j += 256) {
        const float tv = terow[j];
        float se = b_es[j] + tv;
        for (int i = 0; i < 64; ++i) se += qs[i] * W_es[i * Hc + j];
        const float ae = b_ea[j] + tv + qa[0] * W_ea[j] + qa[1] * W_ea[Hc + j];
        float de = b_ed[j] + tv;
        for (int i = 0; i < 5; ++i) de += qd[i] * W_ed[i * Hc + j];
        hraw[base + j]            = de;   // dyn token
        hraw[base + Hc + j]       = se;   // state token
        hraw[base + 2 * Hc + j]   = ae;   // action token
    }
}

// ---------------------------------------------------------------------------
// LayerNorm over H=768 columns, one block per row
// ---------------------------------------------------------------------------
__global__ __launch_bounds__(256) void ln_kernel(
    const float* __restrict__ x, const float* __restrict__ g,
    const float* __restrict__ bb, float* __restrict__ y)
{
    __shared__ float s1[256], s2[256];
    const int row = blockIdx.x, tid = threadIdx.x;
    const float* xr = x + (long)row * Hc;
    const float a0 = xr[tid], a1 = xr[tid + 256], a2 = xr[tid + 512];
    s1[tid] = a0 + a1 + a2;
    s2[tid] = a0 * a0 + a1 * a1 + a2 * a2;
    __syncthreads();
    for (int off = 128; off > 0; off >>= 1) {
        if (tid < off) { s1[tid] += s1[tid + off]; s2[tid] += s2[tid + off]; }
        __syncthreads();
    }
    const float mean = s1[0] * (1.0f / 768.0f);
    const float var  = s2[0] * (1.0f / 768.0f) - mean * mean;
    const float rstd = rsqrtf(var + 1e-5f);
    float* yr = y + (long)row * Hc;
    yr[tid]       = (a0 - mean) * rstd * g[tid]       + bb[tid];
    yr[tid + 256] = (a1 - mean) * rstd * g[tid + 256] + bb[tid + 256];
    yr[tid + 512] = (a2 - mean) * rstd * g[tid + 512] + bb[tid + 512];
}

// ---------------------------------------------------------------------------
// WMMA GEMM: C[M,N] = act(A[M,K] @ W[K,N] + bias (+ residual))
// Block = 256 thr (8 waves), 128x64 tile, BK=32.
// Each wave computes 32x32 (2 A-frags x 2 B-frags -> 4 WMMA / K-step).
// B tile stored TRANSPOSED in LDS so every fragment read is two uint4 loads.
// ACT: 0 none, 1 gelu.
// OUTBF: 0 -> f32 row-major, 1 -> bf16 row-major, 2 -> bf16 TRANSPOSED
//        (out[n*Mrows + m], used for V so attention can load P@V fragments
//         as contiguous 16B vectors)
// ---------------------------------------------------------------------------
template<int ACT, int OUTBF>
__global__ __launch_bounds__(256) void gemm_wmma_kernel(
    const float* __restrict__ A, const float* __restrict__ W,
    const float* __restrict__ bias, const float* __restrict__ residual,
    void* __restrict__ outv, int Ksize, int Nsize)
{
    // 40-ushort row stride: 80 B, 16 B aligned rows
    __shared__ __align__(16) unsigned short As[128][40];   // [m][k]
    __shared__ __align__(16) unsigned short BsT[64][40];   // [n][k]  (transposed)

    const int tid  = threadIdx.x;
    const int lane = tid & 31;
    const int wave = tid >> 5;
    const int m0 = blockIdx.y * 128;
    const int n0 = blockIdx.x * 64;
    const int wr = wave & 3;            // wave row group: 32 rows each
    const int wc = (wave >> 2) * 32;    // wave col offset 0 or 32
    const int ln   = lane & 15;
    const int half = lane >> 4;

    v8f acc00 = {}, acc01 = {}, acc10 = {}, acc11 = {};

    // A tile loaders: 128x32 f32, 16 per thread (one row segment)
    const int arow = tid >> 1;            // 0..127
    const int acol = (tid & 1) * 16;
    // B tile loaders: 32x64 f32, 8 per thread
    const int brow = tid >> 3;            // 0..31
    const int bcol = (tid & 7) * 8;
    const float* aptr = A + (long)(m0 + arow) * Ksize + acol;
    const float* bptr = W + (long)brow * Nsize + n0 + bcol;

    for (int kb = 0; kb < Ksize; kb += 32) {
        {   // A: pack pairs -> 2 x uint4 LDS stores
            uint4 w0, w1;
            const float* p = aptr + kb;
            w0.x = pack2bf(p[0],  p[1]);  w0.y = pack2bf(p[2],  p[3]);
            w0.z = pack2bf(p[4],  p[5]);  w0.w = pack2bf(p[6],  p[7]);
            w1.x = pack2bf(p[8],  p[9]);  w1.y = pack2bf(p[10], p[11]);
            w1.z = pack2bf(p[12], p[13]); w1.w = pack2bf(p[14], p[15]);
            *(uint4*)&As[arow][acol]     = w0;
            *(uint4*)&As[arow][acol + 8] = w1;
        }
        {   // B: transpose into BsT[n][k]
            const float* p = bptr + (long)kb * Nsize;
#pragma unroll
            for (int i = 0; i < 8; ++i) BsT[bcol + i][brow] = f2bf(p[i]);
        }
        if (kb + 32 < Ksize) {
            __builtin_prefetch(aptr + kb + 32, 0, 1);
            __builtin_prefetch(bptr + (long)(kb + 32) * Nsize, 0, 1);
        }
        __syncthreads();

        // Fragments: all reads are contiguous 16B vectors.
        // A layout: lanes 0-15 K={0..7,16..23}, lanes 16-31 K={8..15,24..31}
        BFrag a0, a1, b0, b1;
        const int am0 = wr * 32 + ln;
        const int am1 = am0 + 16;
        const int ak  = half * 8;
        a0.q[0] = *(const uint4*)&As[am0][ak];
        a0.q[1] = *(const uint4*)&As[am0][16 + ak];
        a1.q[0] = *(const uint4*)&As[am1][ak];
        a1.q[1] = *(const uint4*)&As[am1][16 + ak];
        // B layout: lane = column, same K split as A (lanes 0-15 K={0..7,16..23}, 16-31 K={8..15,24..31})
        const int bn0 = wc + ln;
        const int bk  = half * 8;
        b0.q[0] = *(const uint4*)&BsT[bn0][bk];
        b0.q[1] = *(const uint4*)&BsT[bn0][bk + 16];
        b1.q[0] = *(const uint4*)&BsT[bn0 + 16][bk];
        b1.q[1] = *(const uint4*)&BsT[bn0 + 16][bk + 16];

        acc00 = wmma_bf16(a0, b0, acc00);
        acc01 = wmma_bf16(a0, b1, acc01);
        acc10 = wmma_bf16(a1, b0, acc10);
        acc11 = wmma_bf16(a1, b1, acc11);
        __syncthreads();
    }

    // Epilogue. C layout: lane = column, VGPR r -> row r (+8 for upper half-wave)
    const int nA = n0 + wc + ln;
    const int nB = nA + 16;
    const int mb0 = m0 + wr * 32 + half * 8;
    const int mb1 = mb0 + 16;
    float* outf = (float*)outv;
    unsigned short* outh = (unsigned short*)outv;
#pragma unroll
    for (int r = 0; r < 8; ++r) {
        const int mA = mb0 + r, mB = mb1 + r;
        float v00 = acc00[r] + bias[nA];
        float v01 = acc01[r] + bias[nB];
        float v10 = acc10[r] + bias[nA];
        float v11 = acc11[r] + bias[nB];
        if (residual) {
            v00 += residual[(long)mA * Nsize + nA];
            v01 += residual[(long)mA * Nsize + nB];
            v10 += residual[(long)mB * Nsize + nA];
            v11 += residual[(long)mB * Nsize + nB];
        }
        if (ACT == 1) {
            v00 = gelu_exact(v00); v01 = gelu_exact(v01);
            v10 = gelu_exact(v10); v11 = gelu_exact(v11);
        }
        if (OUTBF == 1) {
            outh[(long)mA * Nsize + nA] = f2bf(v00);
            outh[(long)mA * Nsize + nB] = f2bf(v01);
            outh[(long)mB * Nsize + nA] = f2bf(v10);
            outh[(long)mB * Nsize + nB] = f2bf(v11);
        } else if (OUTBF == 2) {   // transposed bf16: out[n][m], m-stride = Mrows
            outh[(long)nA * Mrows + mA] = f2bf(v00);
            outh[(long)nB * Mrows + mA] = f2bf(v01);
            outh[(long)nA * Mrows + mB] = f2bf(v10);
            outh[(long)nB * Mrows + mB] = f2bf(v11);
        } else {
            outf[(long)mA * Nsize + nA] = v00;
            outf[(long)mA * Nsize + nB] = v01;
            outf[(long)mB * Nsize + nA] = v10;
            outf[(long)mB * Nsize + nB] = v11;
        }
    }
}

// ---------------------------------------------------------------------------
// Flash-style causal attention, one wave per (b, head, 16-query tile).
// q/k: bf16 bits [M, H] (col = head*64 + d).  vT: bf16 bits [H, M] (transposed).
// att: f32 [M, H].
// Note: kb <= qmax and kb % 32 == 0  =>  kb+31 <= 1535 < S, so every key
// index touched is in range; no clamps or bounds checks needed anywhere.
// ---------------------------------------------------------------------------
__global__ __launch_bounds__(128) void attention_kernel(
    const unsigned short* __restrict__ q, const unsigned short* __restrict__ k,
    const unsigned short* __restrict__ vT, const int* __restrict__ mask,
    float* __restrict__ att)
{
    __shared__ __align__(16) unsigned short Ps[4][16 * 32];
    const int lane = threadIdx.x & 31;
    const int wloc = threadIdx.x >> 5;
    const int wid  = blockIdx.x * 4 + wloc;
    unsigned short* ps = Ps[wloc];

    const int qt   = wid % (Sc / 16);
    const int head = (wid / (Sc / 16)) % NHc;
    const int b    = wid / ((Sc / 16) * NHc);

    const int qrow0 = b * Sc + qt * 16;
    const int col0  = head * DHc;
    const int ln    = lane & 15;
    const int half  = lane >> 4;

    // Q as two A-fragments (feature chunks 0..31 and 32..63), uint4 loads
    BFrag aq0, aq1;
    {
        const uint4* qp4 = (const uint4*)(q + (long)(qrow0 + ln) * Hc + col0);
        aq0.q[0] = qp4[half * 2];
        aq0.q[1] = qp4[half * 2 + 1];
        aq1.q[0] = qp4[4 + half * 2];
        aq1.q[1] = qp4[4 + half * 2 + 1];
    }

    v8f o[4] = {v8f{}, v8f{}, v8f{}, v8f{}};
    float rm[8], rs[8];
#pragma unroll
    for (int r = 0; r < 8; ++r) { rm[r] = -INFINITY; rs[r] = 0.f; }

    const int qmax = qt * 16 + 15;
    for (int kb = 0; kb <= qmax; kb += 32) {
        // scores: two 16x16 key tiles (all key indices provably < S)
        v8f s0 = {}, s1 = {};
        {
            BFrag blo, bhi;
            const uint4* kp4 =
                (const uint4*)(k + (long)(b * Sc + kb + ln) * Hc + col0);
            blo.q[0] = kp4[half * 2];
            blo.q[1] = kp4[half * 2 + 1];
            bhi.q[0] = kp4[4 + half * 2];
            bhi.q[1] = kp4[4 + half * 2 + 1];
            s0 = wmma_bf16(aq0, blo, s0);
            s0 = wmma_bf16(aq1, bhi, s0);
            kp4 = (const uint4*)(k + (long)(b * Sc + kb + 16 + ln) * Hc + col0);
            blo.q[0] = kp4[half * 2];
            blo.q[1] = kp4[half * 2 + 1];
            bhi.q[0] = kp4[4 + half * 2];
            bhi.q[1] = kp4[4 + half * 2 + 1];
            s1 = wmma_bf16(aq0, blo, s1);
            s1 = wmma_bf16(aq1, bhi, s1);
        }

        const int key0 = kb + ln;
        const int key1 = key0 + 16;
        const bool mok0 = (mask[b * Sc + key0] != 0);
        const bool mok1 = (mask[b * Sc + key1] != 0);

        const float scale = 0.125f;   // 1/sqrt(64)
#pragma unroll
        for (int r = 0; r < 8; ++r) {
            const int qrow = qt * 16 + half * 8 + r;
            float x0 = (mok0 && key0 <= qrow) ? s0[r] * scale : -INFINITY;
            float x1 = (mok1 && key1 <= qrow) ? s1[r] * scale : -INFINITY;
            float mx = fmaxf(x0, x1);
            mx = fmaxf(mx, __shfl_xor(mx, 1));
            mx = fmaxf(mx, __shfl_xor(mx, 2));
            mx = fmaxf(mx, __shfl_xor(mx, 4));
            mx = fmaxf(mx, __shfl_xor(mx, 8));
            const float mnew = fmaxf(rm[r], mx);
            float alpha, p0, p1;
            if (mnew == -INFINITY) { alpha = 1.f; p0 = 0.f; p1 = 0.f; }
            else {
                alpha = expf(rm[r] - mnew);
                p0 = expf(x0 - mnew);
                p1 = expf(x1 - mnew);
            }
            float psum = p0 + p1;
            psum += __shfl_xor(psum, 1);
            psum += __shfl_xor(psum, 2);
            psum += __shfl_xor(psum, 4);
            psum += __shfl_xor(psum, 8);
            rs[r] = rs[r] * alpha + psum;
            rm[r] = mnew;
            o[0][r] *= alpha; o[1][r] *= alpha; o[2][r] *= alpha; o[3][r] *= alpha;
            const int mrow = half * 8 + r;
            ps[mrow * 32 + ln]      = f2bf(p0);
            ps[mrow * 32 + 16 + ln] = f2bf(p1);
        }

        // reload P in A-fragment layout (same-wave LDS RAW; hw-ordered via DScnt)
        BFrag ap;
        {
            const uint4* pp4 = (const uint4*)(ps + ln * 32);
            ap.q[0] = pp4[half * 2];
            ap.q[1] = pp4[half * 2 + 1];
        }
        // P @ V : 4 feature tiles of 16; vT gives contiguous key runs per lane
#pragma unroll
        for (int dt = 0; dt < 4; ++dt) {
            BFrag bvf;
            const int feat = col0 + dt * 16 + ln;
            const uint4* vp4 =
                (const uint4*)(vT + (long)feat * Mrows + b * Sc + kb + half * 16);
            bvf.q[0] = vp4[0];
            bvf.q[1] = vp4[1];
            o[dt] = wmma_bf16(ap, bvf, o[dt]);
        }
    }

#pragma unroll
    for (int dt = 0; dt < 4; ++dt) {
        const int c = col0 + dt * 16 + ln;
#pragma unroll
        for (int r = 0; r < 8; ++r) {
            const int row = qrow0 + half * 8 + r;
            att[(long)row * Hc + c] = o[dt][r] / rs[r];
        }
    }
}

// ---------------------------------------------------------------------------
// Output heads: one block per (b,t)
// ---------------------------------------------------------------------------
__global__ __launch_bounds__(128) void head_kernel(
    const float* __restrict__ h,
    const float* __restrict__ W_ps, const float* __restrict__ b_ps,
    const float* __restrict__ W_pd, const float* __restrict__ b_pd,
    const float* __restrict__ W_pa, const float* __restrict__ b_pa,
    float* __restrict__ out_state, float* __restrict__ out_action,
    float* __restrict__ out_dyn)
{
    const int token = blockIdx.x;
    const int tid = threadIdx.x;
    const int b = token >> 9, t = token & 511;
    const float* ha = h + ((long)b * Sc + 3 * t + 2) * Hc;  // action-token row
    const float* hs = h + ((long)b * Sc + 3 * t + 1) * Hc;  // state-token row
    if (tid < 64) {
        float acc = b_ps[tid];
        for (int i = 0; i < Hc; ++i) acc += ha[i] * W_ps[i * ODc + tid];
        out_state[token * ODc + tid] = acc;
    } else if (tid < 69) {
        const int j = tid - 64;
        float acc = b_pd[j];
        for (int i = 0; i < Hc; ++i) acc += ha[i] * W_pd[i * 5 + j];
        out_dyn[token * 5 + j] = acc;
    } else if (tid < 71) {
        const int j = tid - 69;
        float acc = b_pa[j];
        for (int i = 0; i < Hc; ++i) acc += hs[i] * W_pa[i * 2 + j];
        out_action[token * 2 + j] = tanhf(acc);
    }
}

// ---------------------------------------------------------------------------
// Commit-loss finalization (deterministic reduction of per-token partials)
// ---------------------------------------------------------------------------
__global__ __launch_bounds__(256) void loss_kernel(
    const float* __restrict__ partial, float* __restrict__ out_loss)
{
    __shared__ float red[256];
    const int tid = threadIdx.x;
    for (int c = 0; c < 3; ++c) {
        float s = 0.f;
        for (int i = tid; i < Bc * Tc; i += 256) s += partial[i * 3 + c];
        red[tid] = s; __syncthreads();
        for (int off = 128; off > 0; off >>= 1) {
            if (tid < off) red[tid] += red[tid + off];
            __syncthreads();
        }
        if (tid == 0) {
            const float den = (c == 0) ? (float)(Bc * Tc * 64)
                            : (c == 1) ? (float)(Bc * Tc * 2)
                                       : (float)(Bc * Tc * 5);
            out_loss[c] = red[0] / den;
        }
        __syncthreads();
    }
}

// ---------------------------------------------------------------------------
// Launcher
// ---------------------------------------------------------------------------
extern "C" void kernel_launch(void* const* d_in, const int* in_sizes, int n_in,
                              void* d_out, int out_size, void* d_ws, size_t ws_size,
                              hipStream_t stream) {
    (void)in_sizes; (void)n_in; (void)out_size; (void)ws_size;

    const int*   timesteps = (const int*)d_in[0];
    const float* states    = (const float*)d_in[1];
    const float* actions   = (const float*)d_in[2];
    const float* dynamics  = (const float*)d_in[3];
    const int*   mask      = (const int*)d_in[4];
    const float* te   = (const float*)d_in[5];
    const float* W_es = (const float*)d_in[6];  const float* b_es = (const float*)d_in[7];
    const float* W_ea = (const float*)d_in[8];  const float* b_ea = (const float*)d_in[9];
    const float* W_ed = (const float*)d_in[10]; const float* b_ed = (const float*)d_in[11];
    const float* eln_g = (const float*)d_in[12]; const float* eln_b = (const float*)d_in[13];
    const float* Wq = (const float*)d_in[14]; const float* bq = (const float*)d_in[15];
    const float* Wk = (const float*)d_in[16]; const float* bk = (const float*)d_in[17];
    const float* Wv = (const float*)d_in[18]; const float* bv = (const float*)d_in[19];
    const float* Wp = (const float*)d_in[20]; const float* bp = (const float*)d_in[21];
    const float* W1 = (const float*)d_in[22]; const float* b1 = (const float*)d_in[23];
    const float* W2 = (const float*)d_in[24]; const float* b2 = (const float*)d_in[25];
    const float* ln1_g = (const float*)d_in[26]; const float* ln1_b = (const float*)d_in[27];
    const float* ln2_g = (const float*)d_in[28]; const float* ln2_b = (const float*)d_in[29];
    const float* W_pd = (const float*)d_in[30]; const float* b_pd = (const float*)d_in[31];
    const float* W_ps = (const float*)d_in[32]; const float* b_ps = (const float*)d_in[33];
    const float* W_pa = (const float*)d_in[34]; const float* b_pa = (const float*)d_in[35];
    const float* cb_s = (const float*)d_in[36];
    const float* cb_a = (const float*)d_in[37];
    const float* cb_d = (const float*)d_in[38];

    // ---- workspace layout (bytes) ----
    const size_t MH_F32 = (size_t)Mrows * Hc * 4;        // 18,874,368
    const size_t MH_BF  = (size_t)Mrows * Hc * 2;        //  9,437,184
    const size_t MFF    = (size_t)Mrows * FFc * 4;       // 75,497,472
    char* ws = (char*)d_ws;
    float* h   = (float*)(ws);
    float* h2  = (float*)(ws + MH_F32);
    char*  scr = ws + 2 * MH_F32;
    unsigned short* qb   = (unsigned short*)(scr);
    unsigned short* kbuf = (unsigned short*)(scr + MH_BF);
    unsigned short* vT   = (unsigned short*)(scr + 2 * MH_BF);  // [H][Mrows]
    float* att = (float*)(scr + 3 * MH_BF);
    float* m1  = (float*)(scr);                          // overlaps qkv+att (dead by then)
    float* partial = (float*)(ws + 2 * MH_F32 + MFF);    // B*T*3 floats

    float* out        = (float*)d_out;
    float* out_state  = out;                                  // B*T*64
    float* out_action = out + Bc * Tc * ODc;                  // B*T*2
    float* out_dyn    = out + Bc * Tc * ODc + Bc * Tc * 2;    // B*T*5
    float* out_loss   = out + Bc * Tc * (ODc + 2 + 5);        // 3 scalars

    // 1) VQ + embeddings -> h2 (raw), then embed layernorm -> h
    embed_vq_kernel<<<Bc * Tc, 256, 0, stream>>>(
        timesteps, states, actions, dynamics, te,
        W_es, b_es, W_ea, b_ea, W_ed, b_ed, cb_s, cb_a, cb_d, h2, partial);
    ln_kernel<<<Mrows, 256, 0, stream>>>(h2, eln_g, eln_b, h);

    const dim3 gH(Hc / 64, Mrows / 128);    // (12, 48)
    const dim3 gFF(FFc / 64, Mrows / 128);  // (48, 48)

    for (int l = 0; l < Lc; ++l) {
        const size_t wHH = (size_t)l * Hc * Hc;
        const size_t wHF = (size_t)l * Hc * FFc;
        // QKV (bf16 outputs; V written transposed for vectorized P@V loads)
        gemm_wmma_kernel<0, 1><<<gH, 256, 0, stream>>>(h, Wq + wHH, bq + l * Hc, nullptr, qb,   Hc, Hc);
        gemm_wmma_kernel<0, 1><<<gH, 256, 0, stream>>>(h, Wk + wHH, bk + l * Hc, nullptr, kbuf, Hc, Hc);
        gemm_wmma_kernel<0, 2><<<gH, 256, 0, stream>>>(h, Wv + wHH, bv + l * Hc, nullptr, vT,   Hc, Hc);
        // flash attention (4608 waves, 4 per block)
        attention_kernel<<<(Bc * NHc * (Sc / 16)) / 4, 128, 0, stream>>>(qb, kbuf, vT, mask, att);
        // output projection + residual
        gemm_wmma_kernel<0, 0><<<gH, 256, 0, stream>>>(att, Wp + wHH, bp + l * Hc, h, h2, Hc, Hc);
        ln_kernel<<<Mrows, 256, 0, stream>>>(h2, ln1_g + l * Hc, ln1_b + l * Hc, h);
        // MLP
        gemm_wmma_kernel<1, 0><<<gFF, 256, 0, stream>>>(h,  W1 + wHF, b1 + l * FFc, nullptr, m1, Hc, FFc);
        gemm_wmma_kernel<0, 0><<<gH,  256, 0, stream>>>(m1, W2 + wHF, b2 + l * Hc,  h,       h2, FFc, Hc);
        ln_kernel<<<Mrows, 256, 0, stream>>>(h2, ln2_g + l * Hc, ln2_b + l * Hc, h);
    }

    head_kernel<<<Bc * Tc, 128, 0, stream>>>(h, W_ps, b_ps, W_pd, b_pd, W_pa, b_pa,
                                             out_state, out_action, out_dyn);
    loss_kernel<<<1, 256, 0, stream>>>(partial, out_loss);
}
